// PTB_SNNDelay_46145128628882
// MI455X (gfx1250) — compile-verified, matrix operand order checked by reading, not run on hardware
//
#include <hip/hip_runtime.h>

// ---------------------------------------------------------------------------
// PTB SNN-Delay network on MI455X (gfx1250), bf16 WMMA implementation.
//
// Shapes: T=100, B=8, EMB=HID=256, VOCAB=10000, K=25, PAD_L=24, PAD_R=12.
// Conv-as-shift-GEMM:  y[n, co] = sum_k sum_c Kd[k][co][c] * Xpad[n + 8k][c]
// where n = t*8 + b and Xpad is stored [time*8 + b][c] (bf16, zero-padded).
//
// GEMM staging uses CDNA5 async global->LDS copies (ASYNCcnt) with LDS
// double-buffering when the toolchain exposes the builtins; otherwise it
// falls back to synchronous uint4 staging.
// ---------------------------------------------------------------------------

typedef __attribute__((ext_vector_type(16))) __bf16 v16bf;
typedef __attribute__((ext_vector_type(8)))  float  v8f;
typedef int v4i __attribute__((vector_size(16)));   // matches builtin "V4i"

#define CIN      256
#define KTAPS    25
#define TILE_M   128
#define TILE_N   128
#define TILE_C   32
#define NSTEPS   (KTAPS * (CIN / TILE_C))   // 200 staged tiles

#if defined(__has_builtin)
#if __has_builtin(__builtin_amdgcn_global_load_async_to_lds_b128) && \
    __has_builtin(__builtin_amdgcn_s_wait_asynccnt)
#define USE_ASYNC_LDS 1
#endif
#endif
#ifndef USE_ASYNC_LDS
#define USE_ASYNC_LDS 0
#endif

#if USE_ASYNC_LDS
using g_v4i = __attribute__((address_space(1))) v4i;   // global
using l_v4i = __attribute__((address_space(3))) v4i;   // LDS
__device__ __forceinline__ void async_cp16(__bf16* ldst, const __bf16* gsrc) {
    // GLOBAL_LOAD_ASYNC_TO_LDS_B128: 16B per lane, tracked by ASYNCcnt
    __builtin_amdgcn_global_load_async_to_lds_b128(
        (g_v4i*)(uintptr_t)gsrc, (l_v4i*)(uintptr_t)ldst, 0, 0);
}
#endif

// ---------------- embedding -> bf16 padded activation buffer ---------------
// xp1: [136 time][8 b][256 c]; valid time window is [24, 124)
__global__ void k_embed(const int* __restrict__ idx,
                        const float* __restrict__ emb,
                        __bf16* __restrict__ xp) {
    int r = blockIdx.x;            // flattened (time*8 + b), 0..1087
    int c = threadIdx.x;           // 0..255
    int time = r >> 3, b = r & 7;
    float v = 0.0f;
    if (time >= 24 && time < 124) {
        int tok = idx[(time - 24) * 8 + b];
        v = emb[(size_t)tok * 256 + c];
    }
    xp[(size_t)r * 256 + c] = (__bf16)v;
}

// ---------------- zero fill (uint granularity) -----------------------------
__global__ void k_zero(unsigned int* __restrict__ p, int n) {
    int i = blockIdx.x * blockDim.x + threadIdx.x;
    if (i < n) p[i] = 0u;
}

// ---------------- dense gaussian kernel materialization --------------------
// Kd layout: [k][co (padded to Mpad)][cin] bf16
__global__ void k_gauss(const float* __restrict__ W,
                        const float* __restrict__ P,
                        const float* __restrict__ S,
                        __bf16* __restrict__ Kd, int M, int Mpad) {
    int co = blockIdx.x;           // 0..Mpad-1
    int ci = threadIdx.x;          // 0..255
    size_t base = (size_t)co * CIN + ci;
    size_t ks = (size_t)Mpad * CIN;
    if (co >= M) {
        for (int k = 0; k < KTAPS; ++k) Kd[(size_t)k * ks + base] = (__bf16)0.0f;
        return;
    }
    size_t pi = (size_t)co * CIN + ci;
    float w = W[pi], p = P[pi], s = S[pi];
    p = fminf(fmaxf(p, -12.0f), 12.0f) + 12.0f;
    float sig  = fabsf(s) + 1.0f;
    float inv  = 1.0f / sig;
    float g[KTAPS];
    float sum = 1e-7f;
    #pragma unroll
    for (int k = 0; k < KTAPS; ++k) {
        float d = ((float)k - p) * inv;
        g[k] = __expf(-0.5f * d * d);
        sum += g[k];
    }
    float r = w / sum;
    #pragma unroll
    for (int k = 0; k < KTAPS; ++k)
        Kd[(size_t)k * ks + base] = (__bf16)(g[k] * r);
}

// ---------------- shift-GEMM with bf16 WMMA --------------------------------
// Y[n][m] = sum_k sum_c Kd[k][m0+m][c] * Xp[n + 8k][c], tile 128x128 / block.
__global__ __launch_bounds__(256)
void k_gemm(const __bf16* __restrict__ Kd,
            const __bf16* __restrict__ Xp,
            float* __restrict__ Y, int Mpad) {
    __shared__ __align__(16) __bf16 lA[2][TILE_M * TILE_C];
    __shared__ __align__(16) __bf16 lB[2][TILE_N * TILE_C];

    const int tid  = threadIdx.x;
    const int wave = tid >> 5;          // wave32: 8 waves / block
    const int lane = tid & 31;
    const int m0 = blockIdx.x * TILE_M;
    const int n0 = blockIdx.y * TILE_N;
    const size_t KS = (size_t)Mpad * CIN;

    const int mw = (wave & 3) * 32;     // 4 waves across M (2 subtiles each)
    const int nw = (wave >> 2) * 64;    // 2 waves across N (4 subtiles each)

    v8f acc[2][4];
    const v8f vzero = {0.f, 0.f, 0.f, 0.f, 0.f, 0.f, 0.f, 0.f};
    #pragma unroll
    for (int mi = 0; mi < 2; ++mi)
        #pragma unroll
        for (int ni = 0; ni < 4; ++ni) acc[mi][ni] = vzero;

    // ISA 16-bit A/B fragment layout: lanes 0-15 take k = 0..7 & 16..23,
    // lanes 16-31 take k = 8..15 & 24..31 of the 32-wide contraction chunk.
    const int frow = lane & 15;
    const int kb   = (lane >> 4) << 3;

    // stage tile `it` (k = it/8, cin block = it%8) into LDS buffer `buf`
    auto stage = [&](int it, int buf) {
        const int k  = it >> 3;
        const int c0 = (it & 7) * TILE_C;
        const __bf16* Asrc = Kd + (size_t)k * KS + (size_t)m0 * CIN + c0;
        const __bf16* Bsrc = Xp + (size_t)(n0 + 8 * k) * CIN + c0;
        #pragma unroll
        for (int s = 0; s < 2; ++s) {
            int cid = s * 256 + tid;         // 0..511
            int row = cid >> 2;              // 0..127
            int col = (cid & 3) * 8;         // 0,8,16,24
#if USE_ASYNC_LDS
            async_cp16(&lA[buf][row * TILE_C + col], &Asrc[(size_t)row * CIN + col]);
            async_cp16(&lB[buf][row * TILE_C + col], &Bsrc[(size_t)row * CIN + col]);
#else
            *(uint4*)&lA[buf][row * TILE_C + col] =
                *(const uint4*)&Asrc[(size_t)row * CIN + col];
            *(uint4*)&lB[buf][row * TILE_C + col] =
                *(const uint4*)&Bsrc[(size_t)row * CIN + col];
#endif
        }
        if ((it & 7) == 0 && k < KTAPS - 1 && tid < 64)
            __builtin_prefetch(Asrc + KS, 0, 1);   // pull next k-slice to cache
    };

    stage(0, 0);
#if USE_ASYNC_LDS
    __builtin_amdgcn_s_wait_asynccnt(0);
#endif
    __syncthreads();

    int buf = 0;
    for (int it = 0; it < NSTEPS; ++it) {
        if (it + 1 < NSTEPS)
            stage(it + 1, buf ^ 1);       // overlaps with WMMA below

        v16bf af[2], bfv[4];
        #pragma unroll
        for (int i = 0; i < 2; ++i) {
            const __bf16* p = &lA[buf][(mw + i * 16 + frow) * TILE_C + kb];
            #pragma unroll
            for (int j = 0; j < 8; ++j) { af[i][j] = p[j]; af[i][8 + j] = p[16 + j]; }
        }
        #pragma unroll
        for (int i = 0; i < 4; ++i) {
            const __bf16* p = &lB[buf][(nw + i * 16 + frow) * TILE_C + kb];
            #pragma unroll
            for (int j = 0; j < 8; ++j) { bfv[i][j] = p[j]; bfv[i][8 + j] = p[16 + j]; }
        }
        #pragma unroll
        for (int mi = 0; mi < 2; ++mi)
            #pragma unroll
            for (int ni = 0; ni < 4; ++ni)
                acc[mi][ni] = __builtin_amdgcn_wmma_f32_16x16x32_bf16(
                    false, af[mi], false, bfv[ni],
                    (short)0, acc[mi][ni], false, false);

#if USE_ASYNC_LDS
        __builtin_amdgcn_s_wait_asynccnt(0);
#endif
        __syncthreads();
        buf ^= 1;
    }

    // C/D layout: VGPR j of lane L holds (m = 8*(L>=16) + j, n = L&15)
    #pragma unroll
    for (int mi = 0; mi < 2; ++mi)
        #pragma unroll
        for (int ni = 0; ni < 4; ++ni) {
            int n = n0 + nw + ni * 16 + (lane & 15);
            int m = m0 + mw + mi * 16 + ((lane >> 4) << 3);
            float* dst = Y + (size_t)n * Mpad + m;
            v8f a = acc[mi][ni];
            float4 lo = make_float4(a[0], a[1], a[2], a[3]);
            float4 hi = make_float4(a[4], a[5], a[6], a[7]);
            *(float4*)(dst)     = lo;
            *(float4*)(dst + 4) = hi;
        }
}

// ---------------- eval-BN + spiking LIF scan -> next padded bf16 buffer ----
__global__ void k_bnlif(const float* __restrict__ Y,
                        const float* __restrict__ ga, const float* __restrict__ be,
                        const float* __restrict__ mn, const float* __restrict__ vr,
                        __bf16* __restrict__ xpn, int Tout) {
    int idx = blockIdx.x * blockDim.x + threadIdx.x;   // 0..2047
    int b = idx >> 8, c = idx & 255;
    float sc = ga[c] * rsqrtf(vr[c] + 1e-5f);
    float sh = be[c] - mn[c] * sc;
    float u = 0.0f;
    for (int t = 0; t < Tout; ++t) {
        float x = Y[((size_t)(t * 8 + b)) * 256 + c] * sc + sh;
        u = u * 0.5f + x;
        float s = (u >= 1.0f) ? 1.0f : 0.0f;
        u -= s;                                        // soft reset (thresh=1)
        xpn[((size_t)((24 + t) * 8 + b)) * 256 + c] = (__bf16)s;
    }
}

// ---------------- non-spiking LIF + slice last T, write f32 output ---------
__global__ void k_lif3(const float* __restrict__ Y3, float* __restrict__ out,
                       int Mpad) {
    int idx = blockIdx.x * blockDim.x + threadIdx.x;   // 0..79999
    if (idx >= 8 * 10000) return;
    int co = idx % 10000, b = idx / 10000;
    float u = 0.0f;
    for (int t = 0; t < 136; ++t) {
        u = u * 0.5f + Y3[(size_t)(t * 8 + b) * Mpad + co];
        if (t >= 36)
            out[((size_t)((t - 36) * 8 + b)) * 10000 + co] = u;
    }
}

// ---------------------------------------------------------------------------
extern "C" void kernel_launch(void* const* d_in, const int* in_sizes, int n_in,
                              void* d_out, int out_size, void* d_ws, size_t ws_size,
                              hipStream_t stream) {
    (void)in_sizes; (void)n_in; (void)out_size; (void)ws_size;

    const int*   inputs = (const int*)  d_in[0];
    const float* embw   = (const float*)d_in[1];
    const float* W1 = (const float*)d_in[2],  *P1 = (const float*)d_in[3],  *S1 = (const float*)d_in[4];
    const float* g1 = (const float*)d_in[5],  *b1 = (const float*)d_in[6];
    const float* m1 = (const float*)d_in[7],  *v1 = (const float*)d_in[8];
    const float* W2 = (const float*)d_in[9],  *P2 = (const float*)d_in[10], *S2 = (const float*)d_in[11];
    const float* g2 = (const float*)d_in[12], *b2 = (const float*)d_in[13];
    const float* m2 = (const float*)d_in[14], *v2 = (const float*)d_in[15];
    const float* W3 = (const float*)d_in[16], *P3 = (const float*)d_in[17], *S3 = (const float*)d_in[18];
    float* out = (float*)d_out;

    // workspace carve-up (256B aligned)
    char* ws = (char*)d_ws;
    auto carve = [&](size_t bytes) -> char* {
        char* p = ws;
        ws += (bytes + 255) & ~(size_t)255;
        return p;
    };
    // padded activation buffers: [times][8][256] bf16, fully zero-initialized slack
    const int XT1 = 136, XT2 = 152, XT3 = 168;       // time rows incl. pad+slack
    __bf16* xp1 = (__bf16*)carve((size_t)XT1 * 8 * 256 * 2);
    __bf16* xp2 = (__bf16*)carve((size_t)XT2 * 8 * 256 * 2);
    __bf16* xp3 = (__bf16*)carve((size_t)XT3 * 8 * 256 * 2);
    __bf16* kd1 = (__bf16*)carve((size_t)KTAPS * 256   * 256 * 2);
    __bf16* kd2 = (__bf16*)carve((size_t)KTAPS * 256   * 256 * 2);
    const int MP3 = 10112;                            // 10000 padded to 79*128
    __bf16* kd3 = (__bf16*)carve((size_t)KTAPS * MP3 * 256 * 2);
    float* y1 = (float*)carve((size_t)896  * 256 * 4);   // N1 = 112*8 = 896
    float* y2 = (float*)carve((size_t)1024 * 256 * 4);   // N2pad = 1024 (992 used)
    float* y3 = (float*)carve((size_t)1152 * MP3 * 4);   // N3pad = 1152 (1088 used)

    // 1) embedding -> padded bf16 (zeros outside valid window)
    k_embed<<<dim3(XT1 * 8), dim3(256), 0, stream>>>(inputs, embw, xp1);

    // 2) zero the downstream padded activation buffers
    {
        int w2 = XT2 * 8 * 256 / 2;   // uint words
        int w3 = XT3 * 8 * 256 / 2;
        k_zero<<<dim3((w2 + 255) / 256), dim3(256), 0, stream>>>((unsigned int*)xp2, w2);
        k_zero<<<dim3((w3 + 255) / 256), dim3(256), 0, stream>>>((unsigned int*)xp3, w3);
    }

    // 3) materialize dense gaussian kernels in bf16
    k_gauss<<<dim3(256),  dim3(256), 0, stream>>>(W1, P1, S1, kd1, 256,   256);
    k_gauss<<<dim3(256),  dim3(256), 0, stream>>>(W2, P2, S2, kd2, 256,   256);
    k_gauss<<<dim3(MP3),  dim3(256), 0, stream>>>(W3, P3, S3, kd3, 10000, MP3);

    // 4) layer 1: shift-GEMM (M=256, N=896) + BN + LIF -> xp2
    k_gemm<<<dim3(2, 896 / 128),  dim3(256), 0, stream>>>(kd1, xp1, y1, 256);
    k_bnlif<<<dim3(8), dim3(256), 0, stream>>>(y1, g1, b1, m1, v1, xp2, 112);

    // 5) layer 2: shift-GEMM (M=256, Npad=1024) + BN + LIF -> xp3
    k_gemm<<<dim3(2, 1024 / 128), dim3(256), 0, stream>>>(kd2, xp2, y2, 256);
    k_bnlif<<<dim3(8), dim3(256), 0, stream>>>(y2, g2, b2, m2, v2, xp3, 124);

    // 6) layer 3: shift-GEMM (Mpad=10112, Npad=1152) + non-spiking LIF -> out
    k_gemm<<<dim3(MP3 / 128, 1152 / 128), dim3(256), 0, stream>>>(kd3, xp3, y3, MP3);
    k_lif3<<<dim3((8 * 10000 + 255) / 256), dim3(256), 0, stream>>>(y3, out, MP3);
}
